// CSAttention_33612414059113
// MI455X (gfx1250) — matrix operand order repaired and checked by reading, not compiled, not run on hardware
//
#include <hip/hip_runtime.h>
#include <hip/hip_fp16.h>

typedef __attribute__((ext_vector_type(16))) _Float16 v16h;
typedef __attribute__((ext_vector_type(8)))  _Float16 v8h;
typedef __attribute__((ext_vector_type(8)))  float    v8f;

#define B_DIM 32
#define N_DIM 1024
#define C_DIM 512
#define L_DIM 512
#define M_TOT (B_DIM * N_DIM)   /* 32768 */

// ---------------------------------------------------------------------------
// Fragment loaders for V_WMMA_F32_16X16X32_F16 (wave32), contiguous k order, same for A and B.
// A (16x32 f16): lane m = lane&15; halves 0..15 hold K=kb..kb+15,
//                kb = 16*(lane>=16).
// B (32x16 f16): lane n = lane&15; halves 0..15 hold K=ks..ks+15,
//                ks = 16*(lane>=16).  Source stored as Bsrc[n][k] (k-contig).
// ---------------------------------------------------------------------------
__device__ __forceinline__ v16h load_frag_a(const _Float16* __restrict__ p,
                                            int lda, int lane) {
  int m  = lane & 15;
  int kb = (lane >> 4) << 4;
  const _Float16* q = p + (size_t)m * lda + kb;
  v8h lo = *(const v8h*)(q);
  v8h hi = *(const v8h*)(q + 8);
  v16h r;
#pragma unroll
  for (int i = 0; i < 8; ++i) { r[i] = lo[i]; r[i + 8] = hi[i]; }
  return r;
}

__device__ __forceinline__ v16h load_frag_b(const _Float16* __restrict__ p,
                                            int ldb, int lane) {
  int n  = lane & 15;
  int ks = (lane >> 4) << 4;
  const _Float16* q = p + (size_t)n * ldb + ks;
  v8h lo = *(const v8h*)(q);
  v8h hi = *(const v8h*)(q + 8);
  v16h r;
#pragma unroll
  for (int i = 0; i < 8; ++i) { r[i] = lo[i]; r[i + 8] = hi[i]; }
  return r;
}

#define WMMA_F16(A, Bm, Cacc) \
  __builtin_amdgcn_wmma_f32_16x16x32_f16(false, (A), false, (Bm), (short)0, (Cacc), false, false)

// ---------------------------------------------------------------------------
// Software-pipelined 32x64 GEMM core: acc[rt][j] (+=) A(32xK) * B(Kx64).
// pa: A row-major (lda, K-contiguous); pb: Bsrc[n][k] row-major (ldb).
// Loads for step k+32 are issued before the WMMAs of step k so the scheduler
// can overlap global_load_b128 latency with matrix math.
// ---------------------------------------------------------------------------
__device__ __forceinline__ void gemm_core_32x64(
    const _Float16* __restrict__ pa, int lda,
    const _Float16* __restrict__ pb, int ldb,
    int K, int lane, v8f acc[2][4]) {
  v16h a0 = load_frag_a(pa,                      lda, lane);
  v16h a1 = load_frag_a(pa + (size_t)16 * lda,   lda, lane);
  v16h b0 = load_frag_b(pb,                      ldb, lane);
  v16h b1 = load_frag_b(pb + (size_t)16 * ldb,   ldb, lane);
  v16h b2 = load_frag_b(pb + (size_t)32 * ldb,   ldb, lane);
  v16h b3 = load_frag_b(pb + (size_t)48 * ldb,   ldb, lane);
  for (int k0 = 32; k0 < K; k0 += 32) {
    v16h na0 = load_frag_a(pa + k0,                    lda, lane);
    v16h na1 = load_frag_a(pa + (size_t)16 * lda + k0, lda, lane);
    v16h nb0 = load_frag_b(pb + k0,                    ldb, lane);
    v16h nb1 = load_frag_b(pb + (size_t)16 * ldb + k0, ldb, lane);
    v16h nb2 = load_frag_b(pb + (size_t)32 * ldb + k0, ldb, lane);
    v16h nb3 = load_frag_b(pb + (size_t)48 * ldb + k0, ldb, lane);
    acc[0][0] = WMMA_F16(a0, b0, acc[0][0]);
    acc[0][1] = WMMA_F16(a0, b1, acc[0][1]);
    acc[0][2] = WMMA_F16(a0, b2, acc[0][2]);
    acc[0][3] = WMMA_F16(a0, b3, acc[0][3]);
    acc[1][0] = WMMA_F16(a1, b0, acc[1][0]);
    acc[1][1] = WMMA_F16(a1, b1, acc[1][1]);
    acc[1][2] = WMMA_F16(a1, b2, acc[1][2]);
    acc[1][3] = WMMA_F16(a1, b3, acc[1][3]);
    a0 = na0; a1 = na1; b0 = nb0; b1 = nb1; b2 = nb2; b3 = nb3;
  }
  acc[0][0] = WMMA_F16(a0, b0, acc[0][0]);
  acc[0][1] = WMMA_F16(a0, b1, acc[0][1]);
  acc[0][2] = WMMA_F16(a0, b2, acc[0][2]);
  acc[0][3] = WMMA_F16(a0, b3, acc[0][3]);
  acc[1][0] = WMMA_F16(a1, b0, acc[1][0]);
  acc[1][1] = WMMA_F16(a1, b1, acc[1][1]);
  acc[1][2] = WMMA_F16(a1, b2, acc[1][2]);
  acc[1][3] = WMMA_F16(a1, b3, acc[1][3]);
}

// C/D f32 16x16: vgpr r holds (M = r + 8*(lane>=16), N = lane&15).

// ---------------------------------------------------------------------------
// Precision conversion / transpose staging kernels
// ---------------------------------------------------------------------------
__global__ void __launch_bounds__(256) cvt_f32_f16_kernel(
    const float* __restrict__ src, _Float16* __restrict__ dst, int n) {
  int i = blockIdx.x * 256 + threadIdx.x;
  if (i < n) dst[i] = (_Float16)src[i];
}

// y[B,N,C] f32 -> yt[B,C,N] f16 (write-coalesced)
__global__ void __launch_bounds__(256) transpose_y_kernel(
    const float* __restrict__ y, _Float16* __restrict__ yt) {
  int i = blockIdx.x * 256 + threadIdx.x;      // flat index into yt
  int n = i & (N_DIM - 1);
  int c = (i >> 10) & (C_DIM - 1);
  int b = i >> 19;
  yt[i] = (_Float16)y[((size_t)b * N_DIM + n) * C_DIM + c];
}

// ---------------------------------------------------------------------------
// GEMM 1: qk = x @ W^T + bias  -> q_h (scaled, f16), k_h (f16)
// M=32768, Nout=1024, K=512.  Wave = 32x64 tile -> 16384 waves.
// ---------------------------------------------------------------------------
__global__ void __launch_bounds__(256) gemm_qk_kernel(
    const _Float16* __restrict__ xh, const _Float16* __restrict__ wh,
    const float* __restrict__ bias,
    _Float16* __restrict__ qh, _Float16* __restrict__ kh) {
  int wid  = (blockIdx.x * 256 + threadIdx.x) >> 5;
  int lane = threadIdx.x & 31;
  int row0 = (wid >> 4) * 32;   // 1024 row tiles
  int col0 = (wid & 15) * 64;   // 16 col groups
  v8f acc[2][4] = {};
  gemm_core_32x64(xh + (size_t)row0 * C_DIM, C_DIM,
                  wh + (size_t)col0 * C_DIM, C_DIM, C_DIM, lane, acc);
  int n  = lane & 15;
  int mb = (lane >> 4) << 3;
  const float scale = 0.044194173824159216f;   // 512^-0.5
#pragma unroll
  for (int rt = 0; rt < 2; ++rt)
#pragma unroll
    for (int j = 0; j < 4; ++j) {
      int col = col0 + 16 * j + n;
      float bv = bias[col];
      int rbase = row0 + 16 * rt + mb;
      if (col0 + 16 * j < L_DIM) {             // wave-uniform branch
#pragma unroll
        for (int r = 0; r < 8; ++r)
          qh[(size_t)(rbase + r) * L_DIM + col] =
              (_Float16)((acc[rt][j][r] + bv) * scale);
      } else {
#pragma unroll
        for (int r = 0; r < 8; ++r)
          kh[(size_t)(rbase + r) * L_DIM + (col - L_DIM)] =
              (_Float16)(acc[rt][j][r] + bv);
      }
    }
}

// ---------------------------------------------------------------------------
// GEMM 2: S[b,n,m] = q[b,n,:] . k[b,m,:]   (f16 out). Per batch 1024x1024xK512.
// ---------------------------------------------------------------------------
__global__ void __launch_bounds__(256) gemm_scores_kernel(
    const _Float16* __restrict__ qh, const _Float16* __restrict__ kh,
    _Float16* __restrict__ S) {
  int wid  = (blockIdx.x * 256 + threadIdx.x) >> 5;
  int lane = threadIdx.x & 31;
  int b    = wid >> 9;
  int t    = wid & 511;
  int row0 = (t >> 4) * 32;     // 32 row tiles
  int col0 = (t & 15) * 64;     // 16 col groups
  const _Float16* qb = qh + (size_t)b * N_DIM * L_DIM;
  const _Float16* kb = kh + (size_t)b * N_DIM * L_DIM;
  v8f acc[2][4] = {};
  gemm_core_32x64(qb + (size_t)row0 * L_DIM, L_DIM,
                  kb + (size_t)col0 * L_DIM, L_DIM, L_DIM, lane, acc);
  int n  = lane & 15;
  int mb = (lane >> 4) << 3;
  _Float16* Sb = S + ((size_t)b * N_DIM + row0) * N_DIM;
#pragma unroll
  for (int rt = 0; rt < 2; ++rt)
#pragma unroll
    for (int j = 0; j < 4; ++j)
#pragma unroll
      for (int r = 0; r < 8; ++r)
        Sb[(size_t)(16 * rt + mb + r) * N_DIM + col0 + 16 * j + n] =
            (_Float16)acc[rt][j][r];
}

// ---------------------------------------------------------------------------
// Row softmax over 1024 cols, in place (each block owns exactly one row).
// ---------------------------------------------------------------------------
__global__ void __launch_bounds__(256) softmax_kernel(_Float16* __restrict__ S) {
  __shared__ float red[256];
  _Float16* p = S + (size_t)blockIdx.x * N_DIM;
  int t = threadIdx.x;
  float v[4];
#pragma unroll
  for (int i = 0; i < 4; ++i) v[i] = (float)p[t + 256 * i];
  float mx = fmaxf(fmaxf(v[0], v[1]), fmaxf(v[2], v[3]));
  red[t] = mx; __syncthreads();
  for (int s = 128; s > 0; s >>= 1) {
    if (t < s) red[t] = fmaxf(red[t], red[t + s]);
    __syncthreads();
  }
  mx = red[0]; __syncthreads();
  float sum = 0.f;
#pragma unroll
  for (int i = 0; i < 4; ++i) { v[i] = __expf(v[i] - mx); sum += v[i]; }
  red[t] = sum; __syncthreads();
  for (int s = 128; s > 0; s >>= 1) {
    if (t < s) red[t] += red[t + s];
    __syncthreads();
  }
  float inv = 1.0f / red[0];
#pragma unroll
  for (int i = 0; i < 4; ++i) p[t + 256 * i] = (_Float16)(v[i] * inv);
}

// ---------------------------------------------------------------------------
// GEMM 3: out[b,n,c] = sum_m P[b,n,m] * y_t[b,c,m]   (f32 out, K=1024)
// ---------------------------------------------------------------------------
__global__ void __launch_bounds__(256) gemm_out_kernel(
    const _Float16* __restrict__ P, const _Float16* __restrict__ yt,
    float* __restrict__ out) {
  int wid  = (blockIdx.x * 256 + threadIdx.x) >> 5;
  int lane = threadIdx.x & 31;
  int b    = wid >> 8;
  int t    = wid & 255;
  int row0 = (t >> 3) * 32;     // 32 row tiles
  int col0 = (t & 7) * 64;      // 8 col groups (C=512)
  const _Float16* Pb = P  + (size_t)b * N_DIM * N_DIM;
  const _Float16* yb = yt + (size_t)b * C_DIM * N_DIM;
  v8f acc[2][4] = {};
  gemm_core_32x64(Pb + (size_t)row0 * N_DIM, N_DIM,
                  yb + (size_t)col0 * N_DIM, N_DIM, N_DIM, lane, acc);
  int n  = lane & 15;
  int mb = (lane >> 4) << 3;
  float* ob = out + ((size_t)b * N_DIM + row0) * C_DIM;
#pragma unroll
  for (int rt = 0; rt < 2; ++rt)
#pragma unroll
    for (int j = 0; j < 4; ++j)
#pragma unroll
      for (int r = 0; r < 8; ++r)
        ob[(size_t)(16 * rt + mb + r) * C_DIM + col0 + 16 * j + n] =
            acc[rt][j][r];
}

// ---------------------------------------------------------------------------
extern "C" void kernel_launch(void* const* d_in, const int* in_sizes, int n_in,
                              void* d_out, int out_size, void* d_ws, size_t ws_size,
                              hipStream_t stream) {
  const float* x    = (const float*)d_in[0];   // (32,1024,512)
  const float* y    = (const float*)d_in[1];   // (32,1024,512)
  const float* W    = (const float*)d_in[2];   // (1024,512)
  const float* bias = (const float*)d_in[3];   // (1024,)
  float* out = (float*)d_out;                  // (32,1024,512) f32

  char* ws = (char*)d_ws;
  _Float16* xh = (_Float16*)(ws);                          // 32 MB
  _Float16* wh = (_Float16*)(ws + (size_t)33554432);       //  1 MB
  _Float16* qh = (_Float16*)(ws + (size_t)34603008);       // 32 MB
  _Float16* kh = (_Float16*)(ws + (size_t)68157440);       // 32 MB
  _Float16* yt = (_Float16*)(ws + (size_t)101711872);      // 32 MB
  _Float16* S  = (_Float16*)(ws + (size_t)135266304);      // 64 MB

  cvt_f32_f16_kernel<<<65536, 256, 0, stream>>>(x, xh, M_TOT * C_DIM);
  cvt_f32_f16_kernel<<<2048, 256, 0, stream>>>(W, wh, 2 * L_DIM * C_DIM);
  transpose_y_kernel<<<65536, 256, 0, stream>>>(y, yt);

  gemm_qk_kernel<<<2048, 256, 0, stream>>>(xh, wh, bias, qh, kh);
  gemm_scores_kernel<<<2048, 256, 0, stream>>>(qh, kh, S);
  softmax_kernel<<<32768, 256, 0, stream>>>(S);
  gemm_out_kernel<<<1024, 256, 0, stream>>>(S, yt, out);
}
